// Decoder_30949534335472
// MI455X (gfx1250) — matrix operand order repaired and checked by reading, not compiled, not run on hardware
//
#include <hip/hip_runtime.h>
#include <hip/hip_bf16.h>

#define HDIM 512
#define NTOK 32768           // B*S = 16*2048
#define TM   32              // tokens per workgroup (2 WMMA M-tiles)
#define AS   520             // padded LDS row stride (bf16 elems); 520*2=1040 = 65*16 -> 16B aligned rows

typedef __attribute__((ext_vector_type(16))) __bf16 v16bf;
typedef __attribute__((ext_vector_type(8)))  __bf16 v8bf;
typedef __attribute__((ext_vector_type(4)))  __bf16 v4bf;
typedef __attribute__((ext_vector_type(8)))  float  v8f;

__device__ __forceinline__ v16bf cat8(v8bf lo, v8bf hi) {
    return __builtin_shufflevector(lo, hi, 0,1,2,3,4,5,6,7,8,9,10,11,12,13,14,15);
}

// A-matrix fragment (16x32 bf16): lane holds row m = lane&15.
// v16bf element i -> K = (i<8) ? half*8+i : 16+half*8+(i-8)  => two contiguous 16B chunks.
__device__ __forceinline__ v16bf load_a_frag(const __bf16* row, int kb, int half) {
    const __bf16* p = row + kb + half * 8;
    v8bf lo = *(const v8bf*)p;
    v8bf hi = *(const v8bf*)(p + 16);
    return cat8(lo, hi);
}

// B-matrix fragment (32x16 bf16, Bt rows = weight rows, row-major):
// lane holds column n = lane&15; element i -> K = (i<8) ? half*8+i : 16+half*8+(i-8) => two contiguous 16B chunks.
__device__ __forceinline__ v16bf load_b_frag(const __bf16* row, int kb, int half) {
    const __bf16* p = row + kb + half * 8;
    v8bf lo = *(const v8bf*)p;
    v8bf hi = *(const v8bf*)(p + 16);
    return cat8(lo, hi);
}

__device__ __forceinline__ v8f wmma_bf16(v16bf a, v16bf b, v8f c) {
    return __builtin_amdgcn_wmma_f32_16x16x32_bf16(false, a, false, b, (short)0, c, false, false);
}

// hardware V_TANH_F32 (TRANS32 op on gfx1250) if the toolchain exposes it
__device__ __forceinline__ float fast_tanh(float x) {
#if __has_builtin(__builtin_amdgcn_tanhf)
    return __builtin_amdgcn_tanhf(x);
#else
    return tanhf(x);
#endif
}

// butterfly sum across the 16 lanes sharing the same (half, r)
__device__ __forceinline__ float red16(float v) {
    v += __shfl_xor(v, 8, 32);
    v += __shfl_xor(v, 4, 32);
    v += __shfl_xor(v, 2, 32);
    v += __shfl_xor(v, 1, 32);
    return v;
}

// ---- prep: convert weights to bf16 (and relu / transpose W1) into workspace ----
__global__ void prep_weights(const float* __restrict__ Wh1, const float* __restrict__ W1,
                             __bf16* __restrict__ Whb, __bf16* __restrict__ W1pb,
                             __bf16* __restrict__ W1ptb) {
    int i = blockIdx.x * blockDim.x + threadIdx.x;
    if (i >= HDIM * HDIM) return;
    Whb[i] = (__bf16)Wh1[i];
    float w  = W1[i];
    float wp = w > 0.f ? w : 0.f;
    W1pb[i] = (__bf16)wp;
    int k = i >> 9, hc = i & (HDIM - 1);
    W1ptb[hc * HDIM + k] = (__bf16)wp;      // W1p transposed (rows = h, cols = k)
}

// ---- fused hazard/intensity kernel: one block per 32 tokens, 8 wave32 waves ----
__global__ __launch_bounds__(256) void fused_hazard(
    const float* __restrict__ h,  const float* __restrict__ t,
    const float* __restrict__ b_h1, const float* __restrict__ w_t1,
    const float* __restrict__ b1, const float* __restrict__ W2,
    const float* __restrict__ Wr, const float* __restrict__ b_r,
    const __bf16* __restrict__ Whb, const __bf16* __restrict__ W1pb,
    const __bf16* __restrict__ W1ptb, float* __restrict__ out)
{
    // bufHG: h tile (phase 0/1), then aliased as g = W2p*(1-z^2) tile (phase 2/3)
    __shared__ __align__(16) __bf16 bufHG[TM * AS];
    // bufA rows: [0,32): a0 = tanh(h2) for tokens 0..31 ; [32,64): a = tanh(h2+wt*t)
    __shared__ __align__(16) __bf16 bufA[2 * TM * AS];
    __shared__ float s_wt[HDIM], s_b1p[HDIM], s_w2p[HDIM], s_bh1[HDIM];
    __shared__ float s_t[TM], s_res[TM], s_h0[TM], s_h1[TM], s_int[TM];

    const int tid  = threadIdx.x;
    const int lane = tid & 31;
    const int wave = tid >> 5;
    const int half = lane >> 4;
    const int nlo  = lane & 15;
    const int m0   = blockIdx.x * TM;

    // ---- phase 0a: stage relu'd vectors, init accumulators ----
    for (int k = tid; k < HDIM; k += 256) {
        float a = w_t1[k]; s_wt[k]  = a > 0.f ? a : 0.f;
        float b = b1[k];   s_b1p[k] = b > 0.f ? b : 0.f;
        float c = W2[k];   s_w2p[k] = c > 0.f ? c : 0.f;
        s_bh1[k] = b_h1[k];
    }
    if (tid < TM) {
        s_t[tid]   = t[m0 + tid];
        s_res[tid] = b_r[0];
        s_h0[tid] = 0.f; s_h1[tid] = 0.f; s_int[tid] = 0.f;
    }
    __syncthreads();

    // ---- phase 0b: single pass over h: bf16 tile conversion + res partials ----
    {
        const int m = tid >> 3, part = tid & 7;          // 8 threads per row, 64 elems each
        const float4* hr = (const float4*)(h + (size_t)(m0 + m) * HDIM + part * 64);
        const float4* wr = (const float4*)(Wr + part * 64);
        __bf16* dst = &bufHG[m * AS + part * 64];
        float s = 0.f;
        #pragma unroll
        for (int c = 0; c < 16; ++c) {
            float4 hv = hr[c];
            float4 wv = wr[c];
            s += hv.x * wv.x + hv.y * wv.y + hv.z * wv.z + hv.w * wv.w;
            v4bf pk;
            pk[0] = (__bf16)hv.x; pk[1] = (__bf16)hv.y;
            pk[2] = (__bf16)hv.z; pk[3] = (__bf16)hv.w;
            *(v4bf*)(dst + c * 4) = pk;
        }
        atomicAdd(&s_res[m], s);
    }
    __syncthreads();

    float omaw[4][2][8];   // (1 - a^2) * wt at this lane's (tt, m-tile, r) positions

    // ---- phase 1: GEMM1: h2 = h @ W_h1^T + b_h1 (2 M-tiles share each B frag) ----
    #pragma unroll
    for (int tt = 0; tt < 4; ++tt) {
        const int n = (wave * 4 + tt) * 16 + nlo;
        v8f acc0 = {}, acc1 = {};
        const __bf16* arow0 = &bufHG[nlo * AS];
        const __bf16* arow1 = &bufHG[(16 + nlo) * AS];
        const __bf16* brow  = &Whb[(size_t)n * HDIM];
        #pragma unroll 4
        for (int kb = 0; kb < HDIM; kb += 32) {
            v16bf bf  = load_b_frag(brow,  kb, half);
            v16bf af0 = load_a_frag(arow0, kb, half);
            v16bf af1 = load_a_frag(arow1, kb, half);
            acc0 = wmma_bf16(af0, bf, acc0);
            acc1 = wmma_bf16(af1, bf, acc1);
        }
        const float wtn = s_wt[n];
        #pragma unroll
        for (int mt = 0; mt < 2; ++mt) {
            #pragma unroll
            for (int r = 0; r < 8; ++r) {
                const int m = mt * 16 + r + 8 * half;
                float h2 = (mt ? acc1[r] : acc0[r]) + s_bh1[n];
                float a0 = fast_tanh(h2);
                float av = fast_tanh(h2 + wtn * s_t[m]);
                omaw[tt][mt][r] = (1.f - av * av) * wtn;
                bufA[m * AS + n]        = (__bf16)a0;
                bufA[(TM + m) * AS + n] = (__bf16)av;
            }
        }
    }
    __syncthreads();

    // ---- phase 2: combined GEMM (4 M-tiles share each B frag):
    //      [a0(0:15); a0(16:31); a(0:15); a(16:31)] @ W1p^T + b1p ----
    #pragma unroll
    for (int tt = 0; tt < 4; ++tt) {
        const int n = (wave * 4 + tt) * 16 + nlo;     // n = k' (W1p row)
        v8f acc[4] = {{}, {}, {}, {}};
        const __bf16* brow = &W1pb[(size_t)n * HDIM];
        #pragma unroll 4
        for (int kb = 0; kb < HDIM; kb += 32) {
            v16bf bf = load_b_frag(brow, kb, half);
            #pragma unroll
            for (int j = 0; j < 4; ++j) {
                v16bf af = load_a_frag(&bufA[(j * 16 + nlo) * AS], kb, half);
                acc[j] = wmma_bf16(af, bf, acc[j]);
            }
        }
        const float b1n = s_b1p[n], w2n = s_w2p[n];
        #pragma unroll
        for (int j = 0; j < 4; ++j) {
            #pragma unroll
            for (int r = 0; r < 8; ++r) {
                const int m = (j & 1) * 16 + r + 8 * half;   // token within tile
                float z = fast_tanh(acc[j][r] + b1n);
                float p = red16(w2n * z);
                if (nlo == 0) {
                    if (j < 2) atomicAdd(&s_h0[m], p);
                    else       atomicAdd(&s_h1[m], p);
                }
                if (j >= 2)    // g tile overwrites the dead h tile
                    bufHG[m * AS + n] = (__bf16)(w2n * (1.f - z * z));
            }
        }
    }
    __syncthreads();

    // ---- phase 3: GEMM3: v = g @ W1p (via W1p^T rows); intensity reduce ----
    #pragma unroll
    for (int tt = 0; tt < 4; ++tt) {
        const int n = (wave * 4 + tt) * 16 + nlo;     // n = h index, matches omaw positions
        v8f acc0 = {}, acc1 = {};
        const __bf16* arow0 = &bufHG[nlo * AS];
        const __bf16* arow1 = &bufHG[(16 + nlo) * AS];
        const __bf16* brow  = &W1ptb[(size_t)n * HDIM];
        #pragma unroll 4
        for (int kb = 0; kb < HDIM; kb += 32) {
            v16bf bf  = load_b_frag(brow,  kb, half);
            v16bf af0 = load_a_frag(arow0, kb, half);
            v16bf af1 = load_a_frag(arow1, kb, half);
            acc0 = wmma_bf16(af0, bf, acc0);
            acc1 = wmma_bf16(af1, bf, acc1);
        }
        #pragma unroll
        for (int mt = 0; mt < 2; ++mt) {
            #pragma unroll
            for (int r = 0; r < 8; ++r) {
                const int m = mt * 16 + r + 8 * half;
                float p = red16((mt ? acc1[r] : acc0[r]) * omaw[tt][mt][r]);
                if (nlo == 0) atomicAdd(&s_int[m], p);
            }
        }
    }
    __syncthreads();

    // ---- phase 4: outputs (b2p cancels in both hazard and intensity) ----
    if (tid < TM) {
        const int m = tid;
        float resv = s_res[m]; resv = resv > 0.f ? resv : 0.f;
        float haz   = resv * s_t[m] + s_h1[m] - s_h0[m];
        float inten = resv + s_int[m] + 1e-6f;
        out[m0 + m]        = inten;   // intensity (B,S)
        out[NTOK + m0 + m] = haz;     // hazard (B,S,1)
    }
}

extern "C" void kernel_launch(void* const* d_in, const int* in_sizes, int n_in,
                              void* d_out, int out_size, void* d_ws, size_t ws_size,
                              hipStream_t stream) {
    const float* h    = (const float*)d_in[0];
    const float* t    = (const float*)d_in[1];
    const float* W_h1 = (const float*)d_in[2];
    const float* b_h1 = (const float*)d_in[3];
    const float* w_t1 = (const float*)d_in[4];
    const float* W1   = (const float*)d_in[5];
    const float* b1   = (const float*)d_in[6];
    const float* W2   = (const float*)d_in[7];
    // d_in[8] = b2: cancels analytically; unused
    const float* W_r  = (const float*)d_in[9];
    const float* b_r  = (const float*)d_in[10];

    __bf16* Whb   = (__bf16*)d_ws;            // 512KB
    __bf16* W1pb  = Whb  + HDIM * HDIM;       // 512KB
    __bf16* W1ptb = W1pb + HDIM * HDIM;       // 512KB  (total 1.5MB of ws)

    prep_weights<<<(HDIM * HDIM + 255) / 256, 256, 0, stream>>>(W_h1, W1, Whb, W1pb, W1ptb);
    fused_hazard<<<NTOK / TM, 256, 0, stream>>>(h, t, b_h1, w_t1, b1, W2, W_r, b_r,
                                                Whb, W1pb, W1ptb, (float*)d_out);
}
